// GCN_55482387530472
// MI455X (gfx1250) — matrix operand order repaired and checked by reading, not compiled, not run on hardware
//
#include <hip/hip_runtime.h>

// Problem constants (match reference)
#define NN      12500
#define NP      12544          // NN padded to multiple of 64 (196 stripes)
#define EE      80000
#define BB      8
#define DD      128
#define HH      128
#define NEG_SLOPE 0.2f

typedef __attribute__((ext_vector_type(16))) _Float16 v16h;
typedef __attribute__((ext_vector_type(4)))  _Float16 v4h;
typedef __attribute__((ext_vector_type(8)))  float    v8f;
typedef __attribute__((ext_vector_type(4)))  float    v4f;
typedef __attribute__((ext_vector_type(8)))  unsigned v8u;
typedef __attribute__((ext_vector_type(4)))  unsigned v4u;

// Packed-W element count: 8 col tiles * 4 k-steps * 32 lanes * 16 halves
#define WPACK_ELEMS (8 * 4 * 32 * 16)

// ---------------------------------------------------------------------------
// Repack W1/W2 (f32 [128,128] row-major k->j) into f16 WMMA B-fragment order:
// P[((ct*4 + kt)*32 + lane)*16 + e] = W[(kt*32 + (lane/16)*16 + e)*H + ct*16 + lane%16]
// so each lane's 16-half fragment is one contiguous 32B chunk (2x b128).
__global__ void pack_w_kernel(const float* __restrict__ W1,
                              const float* __restrict__ W2,
                              _Float16* __restrict__ P1,
                              _Float16* __restrict__ P2) {
    int i = blockIdx.x * blockDim.x + threadIdx.x;
    if (i >= WPACK_ELEMS) return;
    int e    = i & 15;
    int lane = (i >> 4) & 31;
    int kt   = (i >> 9) & 3;
    int ct   = (i >> 11) & 7;
    int k = kt * 32 + ((lane >> 4) << 4) + e;
    int j = ct * 16 + (lane & 15);
    P1[i] = (_Float16)W1[k * HH + j];
    P2[i] = (_Float16)W2[k * HH + j];
}

// ---------------------------------------------------------------------------
// h0[n][d4] = emb_atom[atom[n]][d4] + emb_chiral[chiral[n]][d4]  (float4 -> 4xf16)
__global__ void embed_kernel(const int* __restrict__ atom,
                             const int* __restrict__ chiral,
                             const float* __restrict__ emb_atom,
                             const float* __restrict__ emb_chiral,
                             _Float16* __restrict__ h16) {
    int i = blockIdx.x * blockDim.x + threadIdx.x;   // over NN*32 quads
    if (i >= NN * 32) return;
    int n = i >> 5;
    int q = i & 31;
    v4f va = ((const v4f*)(emb_atom   + (size_t)atom[n]   * DD))[q];
    v4f vc = ((const v4f*)(emb_chiral + (size_t)chiral[n] * DD))[q];
    v4f v  = va + vc;
    v4h h;
#pragma unroll
    for (int k = 0; k < 4; ++k) h[k] = (_Float16)v[k];
    ((v4h*)h16)[i] = h;
}

// deg init to 1.0 (self loop weight)
__global__ void deg_init_kernel(float* __restrict__ deg) {
    int n = blockIdx.x * blockDim.x + threadIdx.x;
    if (n < NN) deg[n] = 1.0f;
}

// deg[dst] += 1 per edge
__global__ void deg_acc_kernel(const int* __restrict__ dst, float* __restrict__ deg) {
    int e = blockIdx.x * blockDim.x + threadIdx.x;
    if (e < EE) atomicAdd(&deg[dst[e]], 1.0f);
}

// dis[n] = rsqrt(deg[n])   (in place)
__global__ void deg_rsqrt_kernel(float* __restrict__ deg) {
    int n = blockIdx.x * blockDim.x + threadIdx.x;
    if (n < NN) deg[n] = rsqrtf(deg[n]);
}

// ---------------------------------------------------------------------------
// hw[NP,128] = h16[NP,128] @ W[128,128] via v_wmma_f32_16x16x32_f16.
// All buffers padded to NP rows (multiple of 64) -> no bounds handling at all.
// Each wave owns one 16-col tile, holds its 4 B fragments in VGPRs, and
// sweeps 4 row tiles (64 rows/block). 16 WMMAs per wave.
__global__ void __launch_bounds__(256)
mm_wmma_kernel(const _Float16* __restrict__ A,    // [NP,128] row major
               const _Float16* __restrict__ Bp,   // packed fragments (32KB)
               float* __restrict__ C) {           // [NP,128]
    const int lane    = threadIdx.x & 31;
    const int wave    = threadIdx.x >> 5;      // col tile 0..7
    const int rowBase = blockIdx.x * 64;

    // Prefetch next block's A stripe (global_prefetch_b8; speculative, safe).
    __builtin_prefetch(A + (size_t)(rowBase + 64) * DD + threadIdx.x * 16, 0, 1);

    // ---- B fragments for this column tile, all 4 k-steps (8x b128 total)
    v16h bfrag[4];
#pragma unroll
    for (int kt = 0; kt < 4; ++kt) {
        const v4u* bp = (const v4u*)(Bp + (size_t)(((wave << 2) | kt) * 32 + lane) * 16);
        v4u blo = bp[0];
        v4u bhi = bp[1];
        v8u bu  = __builtin_shufflevector(blo, bhi, 0, 1, 2, 3, 4, 5, 6, 7);
        bfrag[kt] = __builtin_bit_cast(v16h, bu);
    }

    const int nOut = (wave << 4) + (lane & 15);
    const int rOff = (lane >> 4) << 3;         // 0 or 8: C row group of this lane

#pragma unroll
    for (int rt = 0; rt < 4; ++rt) {
        const int m0 = rowBase + rt * 16;
        const _Float16* arow = A + (size_t)(m0 + (lane & 15)) * DD + ((lane >> 4) << 4);

        v8f c = {};
#pragma unroll
        for (int kt = 0; kt < 4; ++kt) {
            // A fragment: dwords {0..7} contiguous from lane k-base -> 2x b128
            const v4u* ap = (const v4u*)(arow + kt * 32);
            v4u alo = ap[0];
            v4u ahi = ap[1];                  // +16 bytes
            v8u au  = __builtin_shufflevector(alo, ahi, 0, 1, 2, 3, 4, 5, 6, 7);
            v16h a  = __builtin_bit_cast(v16h, au);

            c = __builtin_amdgcn_wmma_f32_16x16x32_f16(
                    /*neg_a=*/false, a, /*neg_b=*/false, bfrag[kt],
                    /*c_mod=*/(short)0, c, /*reuse_a=*/false, /*reuse_b=*/false);
        }

        // store C: lane col nOut; VGPR r -> row m0 + rOff + r (stride 512B)
        float* crow = C + (size_t)(m0 + rOff) * HH + nOut;
#pragma unroll
        for (int r = 0; r < 8; ++r) crow[r * HH] = c[r];
    }
}

// ---------------------------------------------------------------------------
// agg[n][d4] = hw[n][d4] * dis[n]^2 + bias[d4]   (self-loop + bias init)
__global__ void agg_init_kernel(const float* __restrict__ hw,
                                const float* __restrict__ dis,
                                const float* __restrict__ bias,
                                float* __restrict__ agg) {
    int i = blockIdx.x * blockDim.x + threadIdx.x;   // over NN*32 quads
    if (i >= NN * 32) return;
    int n = i >> 5;
    int q = i & 31;
    float s = dis[n];
    v4f hv = ((const v4f*)hw)[i];
    v4f bv = ((const v4f*)bias)[q];
    ((v4f*)agg)[i] = hv * (s * s) + bv;
}

// agg[dst] += hw[src] * dis[src]*dis[dst] ; 32 lanes/edge, float4 per lane
__global__ void scatter_kernel(const int* __restrict__ src,
                               const int* __restrict__ dst,
                               const float* __restrict__ hw,
                               const float* __restrict__ dis,
                               float* __restrict__ agg) {
    int idx  = blockIdx.x * blockDim.x + threadIdx.x;
    int e    = idx >> 5;
    int lane = idx & 31;
    if (e >= EE) return;
    int s = src[e], d = dst[e];
    float coef = dis[s] * dis[d];
    v4f msg = ((const v4f*)(hw + (size_t)s * HH))[lane] * coef;
    float* ad = agg + (size_t)d * HH + lane * 4;
#pragma unroll
    for (int i = 0; i < 4; ++i) atomicAdd(&ad[i], msg[i]);
}

// leaky relu in place on agg; also write f16 copy for next layer's GEMM input
__global__ void act_kernel(float* __restrict__ agg, _Float16* __restrict__ h16) {
    int i = blockIdx.x * blockDim.x + threadIdx.x;   // over NN*32 quads
    if (i >= NN * 32) return;
    v4f v = ((const v4f*)agg)[i];
    v4h h;
#pragma unroll
    for (int k = 0; k < 4; ++k) {
        float x = v[k];
        x = (x > 0.0f) ? x : NEG_SLOPE * x;
        v[k] = x;
        h[k] = (_Float16)x;
    }
    ((v4f*)agg)[i] = v;
    ((v4h*)h16)[i] = h;
}

// zero the [B,128] output once
__global__ void out_init_kernel(float* __restrict__ out) {
    int i = blockIdx.x * blockDim.x + threadIdx.x;
    if (i < BB * HH) out[i] = 0.0f;
}

// out[d] += mean_n agg[n][d] ; row-contiguous float4 reads, LDS reduce,
// atomic accumulate into zero-initialized out. 32 blocks.
__global__ void pool_kernel(const float* __restrict__ agg, float* __restrict__ out) {
    __shared__ v4f red[256];
    const int t = threadIdx.x;
    const int q = t & 31;                       // d-slot: d = 4q..4q+3
    v4f acc = {};
    for (int row = blockIdx.x * 8 + (t >> 5); row < NN; row += 32 * 8) {
        acc += ((const v4f*)(agg + (size_t)row * HH))[q];
    }
    red[t] = acc;
    __syncthreads();
    if (t < 32) {
        v4f s = red[t];
#pragma unroll
        for (int w = 1; w < 8; ++w) s += red[t + 32 * w];
        const float scale = 1.0f / (float)NN;
#pragma unroll
        for (int k = 0; k < 4; ++k) atomicAdd(&out[q * 4 + k], s[k] * scale);
    }
}

// ---------------------------------------------------------------------------
extern "C" void kernel_launch(void* const* d_in, const int* in_sizes, int n_in,
                              void* d_out, int out_size, void* d_ws, size_t ws_size,
                              hipStream_t stream) {
    const int*   atom_idx   = (const int*)d_in[0];   // [B,N]
    const int*   chiral_idx = (const int*)d_in[1];   // [B,N]
    const int*   edge_index = (const int*)d_in[2];   // [B,2,E]
    const float* emb_atom   = (const float*)d_in[3]; // [120,128]
    const float* emb_chiral = (const float*)d_in[4]; // [4,128]
    const float* W1         = (const float*)d_in[5]; // [128,128]
    const float* b1         = (const float*)d_in[6]; // [128]
    const float* W2         = (const float*)d_in[7]; // [128,128]
    const float* b2         = (const float*)d_in[8]; // [128]
    float* out = (float*)d_out;                      // [B,128]

    // workspace carve-out (256B aligned); feature buffers padded to NP rows
    char*  ws  = (char*)d_ws;
    size_t off = 0;
    auto take = [&](size_t bytes) -> char* {
        char* p = ws + off;
        off = (off + bytes + 255) & ~(size_t)255;
        return p;
    };
    _Float16* W1p = (_Float16*)take((size_t)WPACK_ELEMS * sizeof(_Float16));
    _Float16* W2p = (_Float16*)take((size_t)WPACK_ELEMS * sizeof(_Float16));
    _Float16* h16 = (_Float16*)take((size_t)NP * HH * sizeof(_Float16));
    float*    hw  = (float*)take((size_t)NP * HH * sizeof(float));
    float*    agg = (float*)take((size_t)NP * HH * sizeof(float));
    float*    dis = (float*)take((size_t)NN * sizeof(float));

    pack_w_kernel<<<(WPACK_ELEMS + 255) / 256, 256, 0, stream>>>(W1, W2, W1p, W2p);
    out_init_kernel<<<(BB * HH + 255) / 256, 256, 0, stream>>>(out);

    const int gridQ   = (NN * 32 + 255) / 256;    // elementwise float4 over [N,128]
    const int gridN   = (NN + 255) / 256;
    const int gridE   = (EE + 255) / 256;
    const int gridE32 = (EE * 32 + 255) / 256;
    const int gridMM  = NP / 64;                  // 196 full stripes of 64 rows

    for (int b = 0; b < BB; ++b) {
        const int* atom   = atom_idx   + (size_t)b * NN;
        const int* chiral = chiral_idx + (size_t)b * NN;
        const int* src    = edge_index + (size_t)b * 2 * EE;
        const int* dst    = src + EE;

        embed_kernel<<<gridQ, 256, 0, stream>>>(atom, chiral, emb_atom, emb_chiral, h16);

        deg_init_kernel<<<gridN, 256, 0, stream>>>(dis);
        deg_acc_kernel<<<gridE, 256, 0, stream>>>(dst, dis);
        deg_rsqrt_kernel<<<gridN, 256, 0, stream>>>(dis);

        // ----- layer 1
        mm_wmma_kernel<<<gridMM, 256, 0, stream>>>(h16, W1p, hw);
        agg_init_kernel<<<gridQ, 256, 0, stream>>>(hw, dis, b1, agg);
        scatter_kernel<<<gridE32, 256, 0, stream>>>(src, dst, hw, dis, agg);
        act_kernel<<<gridQ, 256, 0, stream>>>(agg, h16);

        // ----- layer 2
        mm_wmma_kernel<<<gridMM, 256, 0, stream>>>(h16, W2p, hw);
        agg_init_kernel<<<gridQ, 256, 0, stream>>>(hw, dis, b2, agg);
        scatter_kernel<<<gridE32, 256, 0, stream>>>(src, dst, hw, dis, agg);
        act_kernel<<<gridQ, 256, 0, stream>>>(agg, h16);

        pool_kernel<<<32, 256, 0, stream>>>(agg, out + (size_t)b * HH);
    }
}
